// DenseCL_84035330113833
// MI455X (gfx1250) — matrix operand order repaired and checked by reading, not compiled, not run on hardware
//
#include <hip/hip_runtime.h>
#include <hip/hip_bf16.h>
#include <math.h>

// ---------------------------------------------------------------------------
// Types
// ---------------------------------------------------------------------------
typedef unsigned short u16;
typedef __bf16  bf16x16 __attribute__((ext_vector_type(16)));
typedef float   f32x8   __attribute__((ext_vector_type(8)));
typedef u16     us8     __attribute__((ext_vector_type(8)));
typedef u16     us16    __attribute__((ext_vector_type(16)));
typedef u16     us4     __attribute__((ext_vector_type(4)));
typedef float   f32x4   __attribute__((ext_vector_type(4)));

#define TAUINV 5.0f   // 1/0.2
#define BB 8
#define HW 784
#define MTOT 6272     // B*HW
#define CC 1024
#define DD 2048
#define PP 128
#define QQ 65536

// ---------------------------------------------------------------------------
// Helpers
// ---------------------------------------------------------------------------
__device__ __forceinline__ u16 f2bf(float f) {
    union { float f; unsigned u; } v; v.f = f;
    unsigned u = v.u;
    unsigned r = u + 0x7FFFu + ((u >> 16) & 1u);   // round-to-nearest-even
    return (u16)(r >> 16);
}

// A fragment: 16x32 bf16, per ISA table (half-wave K interleave 0-7/16-23 | 8-15/24-31)
__device__ __forceinline__ bf16x16 frag_a(const u16* p, int lane) {
    int half = (lane & 16) ? 8 : 0;
    us8 lo = *(const us8*)(p + half);
    us8 hi = *(const us8*)(p + 16 + half);
    us16 t = __builtin_shufflevector(lo, hi, 0,1,2,3,4,5,6,7,8,9,10,11,12,13,14,15);
    return __builtin_bit_cast(bf16x16, t);
}

// B fragment: 32x16 bf16, half-wave K interleave 0-7/16-23 | 8-15/24-31 (from Bt row-major [N][K])
__device__ __forceinline__ bf16x16 frag_b(const u16* p, int lane) {
    int off = (lane & 16) ? 8 : 0;
    us8 lo = *(const us8*)(p + off);
    us8 hi = *(const us8*)(p + off + 16);
    us16 t = __builtin_shufflevector(lo, hi, 0,1,2,3,4,5,6,7,8,9,10,11,12,13,14,15);
    return __builtin_bit_cast(bf16x16, t);
}

__device__ __forceinline__ f32x8 wmma_bf16(bf16x16 a, bf16x16 b, f32x8 c) {
    return __builtin_amdgcn_wmma_f32_16x16x32_bf16(false, a, false, b, (short)0, c,
                                                   false, false);
}

__device__ __forceinline__ float dot4(f32x4 a, f32x4 b) {
    return a[0]*b[0] + a[1]*b[1] + a[2]*b[2] + a[3]*b[3];
}

// ---------------------------------------------------------------------------
// Elementwise conversions
// ---------------------------------------------------------------------------
__global__ void k_cvt_bf16(const float* __restrict__ src, u16* __restrict__ dst, int n) {
    int t = blockIdx.x * blockDim.x + threadIdx.x;
    if (t < n) dst[t] = f2bf(src[t]);
}

// W[K][N] f32 -> Wt[N][K] bf16
__global__ void k_transpose_cvt(const float* __restrict__ W, u16* __restrict__ Wt,
                                int K, int N) {
    int t = blockIdx.x * blockDim.x + threadIdx.x;
    if (t < N * K) {
        int n = t / K, k = t % K;
        Wt[t] = f2bf(W[(size_t)k * N + n]);
    }
}

// ---------------------------------------------------------------------------
// Global head (tiny, f32 scalar path)
// ---------------------------------------------------------------------------
__global__ void k_gap(const float* __restrict__ feat, float* __restrict__ g) {
    int t = blockIdx.x * blockDim.x + threadIdx.x;   // [8*1024]
    if (t < BB * CC) {
        int b = t >> 10, c = t & (CC - 1);
        float s = 0.f;
        const float* p = feat + (size_t)b * HW * CC + c;
        for (int i = 0; i < HW; ++i) s += p[(size_t)i * CC];
        g[t] = s * (1.0f / (float)HW);
    }
}

__global__ void k_fc(const float* __restrict__ X, const float* __restrict__ W,
                     const float* __restrict__ bias, float* __restrict__ Y,
                     int Bn, int Cin, int Cout, int do_relu) {
    int t = blockIdx.x * blockDim.x + threadIdx.x;
    if (t < Bn * Cout) {
        int b = t / Cout, o = t % Cout;
        float acc = bias[o];
        const float* x = X + (size_t)b * Cin;
        for (int c = 0; c < Cin; ++c) acc += x[c] * W[(size_t)c * Cout + o];
        if (do_relu) acc = acc > 0.f ? acc : 0.f;
        Y[t] = acc;
    }
}

// ---------------------------------------------------------------------------
// Row L2-normalize (cols = 128), one wave per row; optional bf16 mirror
// ---------------------------------------------------------------------------
__global__ __launch_bounds__(32) void k_l2norm_rows(float* __restrict__ X,
                                                    u16* __restrict__ Xbf) {
    int row = blockIdx.x, lane = threadIdx.x;
    float* p = X + (size_t)row * PP + lane * 4;
    f32x4 v = *(const f32x4*)p;
    float ss = dot4(v, v);
    #pragma unroll
    for (int off = 16; off >= 1; off >>= 1) ss += __shfl_xor(ss, off, 32);
    float s = rsqrtf(fmaxf(ss, 1e-12f));
    v[0] *= s; v[1] *= s; v[2] *= s; v[3] *= s;
    *(f32x4*)p = v;
    if (Xbf) {
        u16* q = Xbf + (size_t)row * PP + lane * 4;
        q[0] = f2bf(v[0]); q[1] = f2bf(v[1]); q[2] = f2bf(v[2]); q[3] = f2bf(v[3]);
    }
}

// ---------------------------------------------------------------------------
// GEMM1: H = relu(A[M,K] @ W1[K,N] + b) -> bf16.
// One wave computes a 32(M) x 64(N) tile: 2 A-frags x 4 B-frags = 8 WMMA / k-step
// (12 b128 loads per 8 WMMAs).
// ---------------------------------------------------------------------------
__global__ __launch_bounds__(32) void k_gemm_relu_bf16(const u16* __restrict__ A,
                                                       const u16* __restrict__ Bt,
                                                       const float* __restrict__ bias,
                                                       u16* __restrict__ H,
                                                       int N, int K) {
    int m0 = blockIdx.x * 32, n0 = blockIdx.y * 64;
    int lane = threadIdx.x;
    const u16* arow0 = A + (size_t)(m0 +  0 + (lane & 15)) * K;
    const u16* arow1 = A + (size_t)(m0 + 16 + (lane & 15)) * K;
    const u16* brow0 = Bt + (size_t)(n0 +  0 + (lane & 15)) * K;
    const u16* brow1 = Bt + (size_t)(n0 + 16 + (lane & 15)) * K;
    const u16* brow2 = Bt + (size_t)(n0 + 32 + (lane & 15)) * K;
    const u16* brow3 = Bt + (size_t)(n0 + 48 + (lane & 15)) * K;
    f32x8 acc[2][4];
    #pragma unroll
    for (int mi = 0; mi < 2; ++mi)
        #pragma unroll
        for (int t = 0; t < 4; ++t) acc[mi][t] = (f32x8){};
    for (int kk = 0; kk < K; kk += 32) {
        bf16x16 a0 = frag_a(arow0 + kk, lane);
        bf16x16 a1 = frag_a(arow1 + kk, lane);
        bf16x16 b0 = frag_b(brow0 + kk, lane);
        bf16x16 b1 = frag_b(brow1 + kk, lane);
        bf16x16 b2 = frag_b(brow2 + kk, lane);
        bf16x16 b3 = frag_b(brow3 + kk, lane);
        acc[0][0] = wmma_bf16(a0, b0, acc[0][0]);
        acc[0][1] = wmma_bf16(a0, b1, acc[0][1]);
        acc[0][2] = wmma_bf16(a0, b2, acc[0][2]);
        acc[0][3] = wmma_bf16(a0, b3, acc[0][3]);
        acc[1][0] = wmma_bf16(a1, b0, acc[1][0]);
        acc[1][1] = wmma_bf16(a1, b1, acc[1][1]);
        acc[1][2] = wmma_bf16(a1, b2, acc[1][2]);
        acc[1][3] = wmma_bf16(a1, b3, acc[1][3]);
    }
    int nloc = lane & 15, hi = (lane >> 4) & 1;
    #pragma unroll
    for (int mi = 0; mi < 2; ++mi) {
        #pragma unroll
        for (int t = 0; t < 4; ++t) {
            int n = n0 + t * 16 + nloc;
            float bv = bias[n];
            #pragma unroll
            for (int r = 0; r < 8; ++r) {
                int m = m0 + mi * 16 + r + hi * 8;
                float v = acc[mi][t][r] + bv;
                v = v > 0.f ? v : 0.f;
                H[(size_t)m * N + n] = f2bf(v);
            }
        }
    }
}

// ---------------------------------------------------------------------------
// GEMM2: OUT = A[M,K] @ W2[K,N] + b -> f32 (no relu). 16(M) x 64(N) tile.
// ---------------------------------------------------------------------------
__global__ __launch_bounds__(32) void k_gemm_f32(const u16* __restrict__ A,
                                                 const u16* __restrict__ Bt,
                                                 const float* __restrict__ bias,
                                                 float* __restrict__ OUT,
                                                 int N, int K) {
    int m0 = blockIdx.x * 16, n0 = blockIdx.y * 64;
    int lane = threadIdx.x;
    const u16* arow = A + (size_t)(m0 + (lane & 15)) * K;
    const u16* brow0 = Bt + (size_t)(n0 +  0 + (lane & 15)) * K;
    const u16* brow1 = Bt + (size_t)(n0 + 16 + (lane & 15)) * K;
    const u16* brow2 = Bt + (size_t)(n0 + 32 + (lane & 15)) * K;
    const u16* brow3 = Bt + (size_t)(n0 + 48 + (lane & 15)) * K;
    f32x8 acc0 = {}, acc1 = {}, acc2 = {}, acc3 = {};
    for (int kk = 0; kk < K; kk += 32) {
        bf16x16 a = frag_a(arow + kk, lane);
        acc0 = wmma_bf16(a, frag_b(brow0 + kk, lane), acc0);
        acc1 = wmma_bf16(a, frag_b(brow1 + kk, lane), acc1);
        acc2 = wmma_bf16(a, frag_b(brow2 + kk, lane), acc2);
        acc3 = wmma_bf16(a, frag_b(brow3 + kk, lane), acc3);
    }
    int nloc = lane & 15, hi = (lane >> 4) & 1;
    f32x8 accs[4] = {acc0, acc1, acc2, acc3};
    #pragma unroll
    for (int t = 0; t < 4; ++t) {
        int n = n0 + t * 16 + nloc;
        float bv = bias[n];
        #pragma unroll
        for (int r = 0; r < 8; ++r) {
            int m = m0 + r + hi * 8;
            OUT[(size_t)m * N + n] = accs[t][r] + bv;
        }
    }
}

// ---------------------------------------------------------------------------
// Dense correspondence: per batch, argmax_n (q_d[m] . k_d[n]); K=128, 784x784
// ---------------------------------------------------------------------------
__global__ __launch_bounds__(32) void k_correspond(const u16* __restrict__ qd,
                                                   const u16* __restrict__ kd,
                                                   int* __restrict__ idxArr) {
    int b = blockIdx.y, m0 = blockIdx.x * 16;
    int lane = threadIdx.x;
    const u16* Ab = qd + (size_t)b * HW * PP;
    const u16* Bb = kd + (size_t)b * HW * PP;
    const u16* arow = Ab + (size_t)(m0 + (lane & 15)) * PP;
    bf16x16 afr[4];
    #pragma unroll
    for (int i = 0; i < 4; ++i) afr[i] = frag_a(arow + i * 32, lane);

    float best_v[8]; int best_n[8];
    #pragma unroll
    for (int r = 0; r < 8; ++r) { best_v[r] = -1e30f; best_n[r] = 0; }

    for (int n0 = 0; n0 < HW; n0 += 16) {
        const u16* brow = Bb + (size_t)(n0 + (lane & 15)) * PP;
        f32x8 acc = {};
        #pragma unroll
        for (int i = 0; i < 4; ++i) acc = wmma_bf16(afr[i], frag_b(brow + i * 32, lane), acc);
        int n = n0 + (lane & 15);
        #pragma unroll
        for (int r = 0; r < 8; ++r) {
            float v = acc[r];
            if (v > best_v[r]) { best_v[r] = v; best_n[r] = n; }
        }
    }
    // tournament across the 16 lanes holding each row (first-max tie-break)
    #pragma unroll
    for (int off = 1; off <= 8; off <<= 1) {
        #pragma unroll
        for (int r = 0; r < 8; ++r) {
            float ov = __shfl_xor(best_v[r], off, 32);
            int   on = __shfl_xor(best_n[r], off, 32);
            if (ov > best_v[r] || (ov == best_v[r] && on < best_n[r])) {
                best_v[r] = ov; best_n[r] = on;
            }
        }
    }
    if ((lane & 15) == 0) {
        int hi = lane >> 4;
        #pragma unroll
        for (int r = 0; r < 8; ++r)
            idxArr[b * HW + m0 + hi * 8 + r] = best_n[r];
    }
}

// matched_bf[row] = kd_bf[b*784 + idx[row]]
__global__ __launch_bounds__(32) void k_gather(const u16* __restrict__ kd,
                                               const int* __restrict__ idxArr,
                                               u16* __restrict__ matched) {
    int row = blockIdx.x, lane = threadIdx.x;
    int b = row / HW;
    int j = idxArr[row];
    const us4* src = (const us4*)(kd + ((size_t)b * HW + j) * PP) + lane;
    us4* dst = (us4*)(matched + (size_t)row * PP) + lane;
    *dst = *src;
}

// ---------------------------------------------------------------------------
// Dense InfoNCE streaming LSE: row_terms[m] = lse_m - logit[m][m]
// fixed max = 1/tau = 5 (rows are unit-norm)
// ---------------------------------------------------------------------------
__global__ __launch_bounds__(32) void k_dense_lse(const u16* __restrict__ qd,
                                                  const u16* __restrict__ matched,
                                                  float* __restrict__ row_terms) {
    int m0 = blockIdx.x * 16, lane = threadIdx.x;
    const u16* arow = qd + (size_t)(m0 + (lane & 15)) * PP;
    bf16x16 afr[4];
    #pragma unroll
    for (int i = 0; i < 4; ++i) afr[i] = frag_a(arow + i * 32, lane);

    float sume[8], pos[8];
    #pragma unroll
    for (int r = 0; r < 8; ++r) { sume[r] = 0.f; pos[r] = 0.f; }
    int hi = (lane >> 4) & 1;

    for (int n0 = 0; n0 < MTOT; n0 += 16) {
        const u16* brow = matched + (size_t)(n0 + (lane & 15)) * PP;
        f32x8 acc = {};
        #pragma unroll
        for (int i = 0; i < 4; ++i) acc = wmma_bf16(afr[i], frag_b(brow + i * 32, lane), acc);
        int n = n0 + (lane & 15);
        #pragma unroll
        for (int r = 0; r < 8; ++r) {
            int m = m0 + r + hi * 8;
            float logit = acc[r] * TAUINV;
            sume[r] += __expf(logit - TAUINV);
            if (n == m) pos[r] = logit;
        }
    }
    #pragma unroll
    for (int off = 1; off <= 8; off <<= 1) {
        #pragma unroll
        for (int r = 0; r < 8; ++r) {
            sume[r] += __shfl_xor(sume[r], off, 32);
            pos[r]  += __shfl_xor(pos[r],  off, 32);   // only one lane nonzero
        }
    }
    if ((lane & 15) == 0) {
        #pragma unroll
        for (int r = 0; r < 8; ++r) {
            int m = m0 + hi * 8 + r;
            row_terms[m] = TAUINV + __logf(sume[r]) - pos[r];
        }
    }
}

// ---------------------------------------------------------------------------
// Queue InfoNCE pieces
// ---------------------------------------------------------------------------
__global__ void k_zero8(float* p) {
    int t = threadIdx.x;
    if (t < 8) p[t] = 0.f;
}

__global__ __launch_bounds__(256) void k_lpos(const float* __restrict__ qg,
                                              const float* __restrict__ kg,
                                              float* __restrict__ lpos) {
    int i = threadIdx.x >> 5, lane = threadIdx.x & 31;
    f32x4 a = *(const f32x4*)(qg + (size_t)i * PP + lane * 4);
    f32x4 b = *(const f32x4*)(kg + (size_t)i * PP + lane * 4);
    float p = dot4(a, b);
    #pragma unroll
    for (int off = 16; off >= 1; off >>= 1) p += __shfl_xor(p, off, 32);
    if (lane == 0) lpos[i] = p;
}

__global__ __launch_bounds__(256) void k_queue_lse(const float* __restrict__ qg,
                                                   const float* __restrict__ queue,
                                                   float* __restrict__ sumexp_g) {
    int tid = blockIdx.x * blockDim.x + threadIdx.x;
    int wid = tid >> 5, lane = tid & 31;
    int nwaves = (gridDim.x * blockDim.x) >> 5;
    f32x4 qreg[BB];
    #pragma unroll
    for (int i = 0; i < BB; ++i) qreg[i] = *(const f32x4*)(qg + (size_t)i * PP + lane * 4);
    float acc[BB];
    #pragma unroll
    for (int i = 0; i < BB; ++i) acc[i] = 0.f;
    for (int j = wid; j < QQ; j += nwaves) {
        f32x4 kv = *(const f32x4*)(queue + (size_t)j * PP + lane * 4);
        #pragma unroll
        for (int i = 0; i < BB; ++i) {
            float p = dot4(qreg[i], kv);
            #pragma unroll
            for (int off = 16; off >= 1; off >>= 1) p += __shfl_xor(p, off, 32);
            acc[i] += __expf(p * TAUINV - TAUINV);
        }
    }
    if (lane == 0) {
        #pragma unroll
        for (int i = 0; i < BB; ++i) atomicAdd(&sumexp_g[i], acc[i]);
    }
}

// ---------------------------------------------------------------------------
// Final reduction: total = 0.5*l_g + 0.5*l_d
// ---------------------------------------------------------------------------
__global__ __launch_bounds__(256) void k_final(const float* __restrict__ row_terms,
                                               const float* __restrict__ sumexp_g,
                                               const float* __restrict__ lpos,
                                               float* __restrict__ out) {
    __shared__ float red[256];
    int t = threadIdx.x;
    float s = 0.f;
    for (int i = t; i < MTOT; i += 256) s += row_terms[i];
    red[t] = s;
    __syncthreads();
    for (int w = 128; w >= 1; w >>= 1) {
        if (t < w) red[t] += red[t + w];
        __syncthreads();
    }
    if (t == 0) {
        float l_d = red[0] * (1.0f / (float)MTOT);
        float l_g = 0.f;
        for (int i = 0; i < BB; ++i) {
            float lp5 = lpos[i] * TAUINV;
            float se = sumexp_g[i] + __expf(lp5 - TAUINV);   // include positive column
            l_g += TAUINV + __logf(se) - lp5;
        }
        l_g *= (1.0f / (float)BB);
        out[0] = 0.5f * l_g + 0.5f * l_d;
    }
}

// ---------------------------------------------------------------------------
// Host: orchestrate the pipeline
// ---------------------------------------------------------------------------
extern "C" void kernel_launch(void* const* d_in, const int* in_sizes, int n_in,
                              void* d_out, int out_size, void* d_ws, size_t ws_size,
                              hipStream_t stream) {
    (void)in_sizes; (void)n_in; (void)out_size; (void)ws_size;
    const float* feat_q = (const float*)d_in[0];
    const float* feat_k = (const float*)d_in[1];
    const float* Wg1 = (const float*)d_in[2];   const float* bg1 = (const float*)d_in[3];
    const float* Wg2 = (const float*)d_in[4];   const float* bg2 = (const float*)d_in[5];
    const float* Wd1 = (const float*)d_in[6];   const float* bd1 = (const float*)d_in[7];
    const float* Wd2 = (const float*)d_in[8];   const float* bd2 = (const float*)d_in[9];
    const float* mWg1 = (const float*)d_in[10]; const float* mbg1 = (const float*)d_in[11];
    const float* mWg2 = (const float*)d_in[12]; const float* mbg2 = (const float*)d_in[13];
    const float* mWd1 = (const float*)d_in[14]; const float* mbd1 = (const float*)d_in[15];
    const float* mWd2 = (const float*)d_in[16]; const float* mbd2 = (const float*)d_in[17];
    const float* queue = (const float*)d_in[18];
    float* out = (float*)d_out;

    char* ws = (char*)d_ws;
    size_t off = 0;
    auto alloc = [&](size_t bytes) -> void* {
        void* p = ws + off;
        off = (off + bytes + 255) & ~(size_t)255;
        return p;
    };
    u16* A_q     = (u16*)alloc((size_t)MTOT * CC * 2);
    u16* A_k     = (u16*)alloc((size_t)MTOT * CC * 2);
    u16* W1t_s   = (u16*)alloc((size_t)DD * CC * 2);
    u16* W1t_m   = (u16*)alloc((size_t)DD * CC * 2);
    u16* W2t_s   = (u16*)alloc((size_t)PP * DD * 2);
    u16* W2t_m   = (u16*)alloc((size_t)PP * DD * 2);
    u16* H_q     = (u16*)alloc((size_t)MTOT * DD * 2);
    u16* H_k     = (u16*)alloc((size_t)MTOT * DD * 2);
    float* qd    = (float*)alloc((size_t)MTOT * PP * 4);
    float* kd    = (float*)alloc((size_t)MTOT * PP * 4);
    u16* qd_bf   = (u16*)alloc((size_t)MTOT * PP * 2);
    u16* kd_bf   = (u16*)alloc((size_t)MTOT * PP * 2);
    u16* mat_bf  = (u16*)alloc((size_t)MTOT * PP * 2);
    int* idxArr  = (int*)alloc((size_t)MTOT * 4);
    float* g_q   = (float*)alloc((size_t)BB * CC * 4);
    float* g_k   = (float*)alloc((size_t)BB * CC * 4);
    float* h_gq  = (float*)alloc((size_t)BB * DD * 4);
    float* h_gk  = (float*)alloc((size_t)BB * DD * 4);
    float* qg    = (float*)alloc((size_t)BB * PP * 4);
    float* kg    = (float*)alloc((size_t)BB * PP * 4);
    float* rowt  = (float*)alloc((size_t)MTOT * 4);
    float* seg   = (float*)alloc(8 * 4);
    float* lpos  = (float*)alloc(8 * 4);

    // --- conversions ---
    {
        int n = MTOT * CC;
        k_cvt_bf16<<<(n + 255) / 256, 256, 0, stream>>>(feat_q, A_q, n);
        k_cvt_bf16<<<(n + 255) / 256, 256, 0, stream>>>(feat_k, A_k, n);
    }
    k_transpose_cvt<<<(DD * CC + 255) / 256, 256, 0, stream>>>(Wd1,  W1t_s, CC, DD);
    k_transpose_cvt<<<(DD * CC + 255) / 256, 256, 0, stream>>>(mWd1, W1t_m, CC, DD);
    k_transpose_cvt<<<(PP * DD + 255) / 256, 256, 0, stream>>>(Wd2,  W2t_s, DD, PP);
    k_transpose_cvt<<<(PP * DD + 255) / 256, 256, 0, stream>>>(mWd2, W2t_m, DD, PP);

    // --- global head (f32 scalar) ---
    k_gap<<<(BB * CC + 255) / 256, 256, 0, stream>>>(feat_q, g_q);
    k_gap<<<(BB * CC + 255) / 256, 256, 0, stream>>>(feat_k, g_k);
    k_fc<<<(BB * DD + 255) / 256, 256, 0, stream>>>(g_q, Wg1,  bg1,  h_gq, BB, CC, DD, 1);
    k_fc<<<(BB * DD + 255) / 256, 256, 0, stream>>>(g_k, mWg1, mbg1, h_gk, BB, CC, DD, 1);
    k_fc<<<(BB * PP + 255) / 256, 256, 0, stream>>>(h_gq, Wg2,  bg2,  qg, BB, DD, PP, 0);
    k_fc<<<(BB * PP + 255) / 256, 256, 0, stream>>>(h_gk, mWg2, mbg2, kg, BB, DD, PP, 0);
    k_l2norm_rows<<<BB, 32, 0, stream>>>(qg, (u16*)nullptr);
    k_l2norm_rows<<<BB, 32, 0, stream>>>(kg, (u16*)nullptr);

    // --- dense heads via WMMA ---
    dim3 g1(MTOT / 32, DD / 64);
    k_gemm_relu_bf16<<<g1, 32, 0, stream>>>(A_q, W1t_s, bd1,  H_q, DD, CC);
    k_gemm_relu_bf16<<<g1, 32, 0, stream>>>(A_k, W1t_m, mbd1, H_k, DD, CC);
    dim3 g2(MTOT / 16, PP / 64);
    k_gemm_f32<<<g2, 32, 0, stream>>>(H_q, W2t_s, bd2,  qd, PP, DD);
    k_gemm_f32<<<g2, 32, 0, stream>>>(H_k, W2t_m, mbd2, kd, PP, DD);
    k_l2norm_rows<<<MTOT, 32, 0, stream>>>(qd, qd_bf);
    k_l2norm_rows<<<MTOT, 32, 0, stream>>>(kd, kd_bf);

    // --- dense correspondence (WMMA + in-register argmax) + gather ---
    dim3 gc(HW / 16, BB);
    k_correspond<<<gc, 32, 0, stream>>>(qd_bf, kd_bf, idxArr);
    k_gather<<<MTOT, 32, 0, stream>>>(kd_bf, idxArr, mat_bf);

    // --- dense InfoNCE streaming LSE (WMMA) ---
    k_dense_lse<<<MTOT / 16, 32, 0, stream>>>(qd_bf, mat_bf, rowt);

    // --- queue InfoNCE ---
    k_zero8<<<1, 32, 0, stream>>>(seg);
    k_lpos<<<1, 256, 0, stream>>>(qg, kg, lpos);
    k_queue_lse<<<512, 256, 0, stream>>>(qg, queue, seg);

    // --- final ---
    k_final<<<1, 256, 0, stream>>>(rowt, seg, lpos, out);
}
